// Encoder_18966575579655
// MI455X (gfx1250) — matrix operand order repaired and checked by reading, not compiled, not run on hardware
//
#include <hip/hip_runtime.h>

// ---------------------------------------------------------------------------
// GraphCast-style encoder for MI455X (gfx1250), bf16 WMMA everywhere.
// GEMM: 4 M-tiles per wave sharing one B fragment; manual ping-pong K
// double-buffer (2x unrolled) -> no inter-buffer copies, loads a full K-tile
// ahead of the consuming v_wmma.
// ---------------------------------------------------------------------------

typedef __attribute__((ext_vector_type(16))) __bf16        v16bf;
typedef __attribute__((ext_vector_type(8)))  float         v8f;
typedef __attribute__((ext_vector_type(4)))  unsigned int  u32x4;

#define NGRID  (360 * 720)          // 259200
#define NH3    5882
#define NNODE  (NGRID + NH3)        // 265082
#define NEDGE  NGRID                // 259200
#define ELAT   (NH3 * 7)            // 41174
#define INDIM  78
#define HID    256

// padded K dims (multiples of 32)
#define INPAD   96
#define EAPAD   32
#define EPPAD   544                 // 2*256+2 = 514 -> 544
#define NPPAD   288                 // 256+2 = 258 -> 288
#define OUT2PAD 16                  // Nout=2 padded to one 16-col tile

#define MT      4                   // M-tiles per wave (register blocking)

__device__ __forceinline__ unsigned short f32_to_bf16(float f) {
    unsigned int u = __float_as_uint(f);
    unsigned int r = u + 0x7FFFu + ((u >> 16) & 1u);
    return (unsigned short)(r >> 16);
}

// Forced global (address_space(1)) 16-byte load -> global_load_b128.
__device__ __forceinline__ u32x4 gload16(const unsigned short* p) {
    typedef __attribute__((address_space(1))) const u32x4* gq;
    return *(gq)(unsigned long long)p;
}

// ---------------------------------------------------------------------------
// Pack weight W[K,N] (row major f32) into WMMA B-fragment order:
// Wp[nt][kt][lane][e] with k = kt*32 + (lane>=16 ? 16:0) + e, n = nt*16 + lane%16
// ---------------------------------------------------------------------------
__global__ void pack_weight_k(const float* __restrict__ W, int K, int N,
                              unsigned short* __restrict__ Wp,
                              int Ktiles, long long total) {
    long long tid = (long long)blockIdx.x * blockDim.x + threadIdx.x;
    if (tid >= total) return;
    int e    = (int)(tid & 15);
    int lane = (int)((tid >> 4) & 31);
    long long rest = tid >> 9;
    int kt = (int)(rest % Ktiles);
    int nt = (int)(rest / Ktiles);
    int k = kt * 32 + ((lane >= 16) ? 16 : 0) + e;
    int n = nt * 16 + (lane & 15);
    float v = (k < K && n < N) ? W[(size_t)k * N + n] : 0.0f;
    Wp[tid] = f32_to_bf16(v);
}

// Cast/pad rows: src f32 [rows,K] -> dst bf16 [rows,Kpad] (zero pad)
__global__ void pack_rows_k(const float* __restrict__ src,
                            unsigned short* __restrict__ dst,
                            int K, int Kpad, long long total) {
    long long tid = (long long)blockIdx.x * blockDim.x + threadIdx.x;
    if (tid >= total) return;
    long long r = tid / Kpad;
    int c = (int)(tid % Kpad);
    float v = (c < K) ? src[r * K + c] : 0.0f;
    dst[tid] = f32_to_bf16(v);
}

// ---------------------------------------------------------------------------
// Generic bf16 WMMA GEMM: Y[M,N] = relu?(A[M,Kpad] * W + bias)
// 8 waves/block; each wave: MT=4 16x16 C tiles sharing one B fragment.
// ---------------------------------------------------------------------------
union FragU { u32x4 q[2]; v16bf v; };

__device__ __forceinline__ void load_tile(FragU& b, FragU (&a)[MT],
                                          const unsigned short* pB,
                                          const unsigned short* const (&pA)[MT],
                                          int kt) {
    const unsigned short* nB = pB + (size_t)kt * 512;
    b.q[0] = gload16(nB);
    b.q[1] = gload16(nB + 8);
#pragma unroll
    for (int t = 0; t < MT; ++t) {
        const unsigned short* nA = pA[t] + kt * 32;
        a[t].q[0] = gload16(nA);        // K = kt*32 + hi*16 + [0..7]
        a[t].q[1] = gload16(nA + 8);    // K = kt*32 + hi*16 + 8 + [0..7]
        __builtin_prefetch(nA + 64, 0, 3);   // 2 K-tiles ahead
    }
}

__device__ __forceinline__ void mma_tile(v8f (&acc)[MT],
                                         const FragU (&a)[MT], const FragU& b) {
#pragma unroll
    for (int t = 0; t < MT; ++t) {
        acc[t] = __builtin_amdgcn_wmma_f32_16x16x32_bf16(
            false, a[t].v, false, b.v, (short)0, acc[t], false, false);
    }
}

__global__ __launch_bounds__(256) void gemm_bf16_k(
    const unsigned short* __restrict__ A, int lda, int M,
    const unsigned short* __restrict__ Wp, int Ktiles,
    const float* __restrict__ bias, int Nout,
    unsigned short* __restrict__ out_bf, int ldobf,
    float* __restrict__ out_f, int ldof, int relu) {

    const int lane  = threadIdx.x & 31;
    const int wave  = threadIdx.x >> 5;
    const int mt0   = (blockIdx.x * 8 + wave) * MT;   // first M-tile of this wave
    const int row0  = mt0 * 16;
    if (row0 >= M) return;

    const int hi = lane >> 4;       // 0/1: K-half select for A, row-half for C
    const int nl = lane & 15;       // A: M index within tile; C: N index

    const unsigned short* pA[MT];
#pragma unroll
    for (int t = 0; t < MT; ++t) {
        int r = row0 + t * 16 + nl;
        if (r >= M) r = M - 1;      // clamp OOB rows; stores guarded below
        pA[t] = A + (size_t)r * lda + hi * 16;
    }
    const unsigned short* pB = Wp + ((size_t)blockIdx.y * Ktiles * 32 + lane) * 16;

    v8f acc[MT] = {};
    FragU b0, b1, a0[MT], a1[MT];

    // prologue: k-tile 0 into buffer 0
    load_tile(b0, a0, pB, pA, 0);

    int kt = 0;
    for (; kt + 2 <= Ktiles; kt += 2) {
        load_tile(b1, a1, pB, pA, kt + 1);      // tile kt+1 -> buf1
        mma_tile(acc, a0, b0);                  // consume tile kt
        if (kt + 2 < Ktiles)
            load_tile(b0, a0, pB, pA, kt + 2);  // tile kt+2 -> buf0
        mma_tile(acc, a1, b1);                  // consume tile kt+1
    }
    if (kt < Ktiles)                            // odd remainder (buf0 holds it)
        mma_tile(acc, a0, b0);

    const int n = blockIdx.y * 16 + nl;
    float bn = (n < Nout) ? bias[n] : 0.0f;
#pragma unroll
    for (int t = 0; t < MT; ++t) {
#pragma unroll
        for (int i = 0; i < 8; ++i) {
            int m = row0 + t * 16 + hi * 8 + i;
            if (m < M && n < Nout) {
                float v = acc[t][i] + bn;
                if (relu) v = fmaxf(v, 0.0f);
                if (out_f)  out_f[(size_t)m * ldof + n] = v;
                if (out_bf) out_bf[(size_t)m * ldobf + n] = f32_to_bf16(v);
            }
        }
    }
}

// ---------------------------------------------------------------------------
// LayerNorm over 256 cols, one wave32 per row; optional residual add and
// bf16 mirror output.
// ---------------------------------------------------------------------------
__global__ __launch_bounds__(256) void ln256_k(
    const float* __restrict__ Y, const float* __restrict__ gamma,
    const float* __restrict__ beta, const float* __restrict__ resid,
    float* __restrict__ out, unsigned short* __restrict__ out_bf, int M) {

    int row  = (int)((blockIdx.x * blockDim.x + threadIdx.x) >> 5);
    int lane = threadIdx.x & 31;
    if (row >= M) return;
    const float* y = Y + (size_t)row * 256;

    float v[8];
    float s = 0.0f;
#pragma unroll
    for (int j = 0; j < 8; ++j) { v[j] = y[lane + 32 * j]; s += v[j]; }
#pragma unroll
    for (int off = 16; off > 0; off >>= 1) s += __shfl_xor(s, off, 32);
    float mu = s * (1.0f / 256.0f);

    float vs = 0.0f;
#pragma unroll
    for (int j = 0; j < 8; ++j) { float d = v[j] - mu; vs += d * d; }
#pragma unroll
    for (int off = 16; off > 0; off >>= 1) vs += __shfl_xor(vs, off, 32);
    float rstd = rsqrtf(vs * (1.0f / 256.0f) + 1e-5f);

#pragma unroll
    for (int j = 0; j < 8; ++j) {
        int c = lane + 32 * j;
        float o = (v[j] - mu) * rstd * gamma[c] + beta[c];
        if (resid) o += resid[(size_t)row * 256 + c];
        out[(size_t)row * 256 + c] = o;
        if (out_bf) out_bf[(size_t)row * 256 + c] = f32_to_bf16(o);
    }
}

// LayerNorm over 2 cols + optional residual, one thread per row.
__global__ void ln2_k(const float* __restrict__ Y, const float* __restrict__ g,
                      const float* __restrict__ b, const float* __restrict__ resid,
                      float* __restrict__ out, int M) {
    int i = blockIdx.x * blockDim.x + threadIdx.x;
    if (i >= M) return;
    float y0 = Y[(size_t)i * 2], y1 = Y[(size_t)i * 2 + 1];
    float mu = 0.5f * (y0 + y1);
    float d0 = y0 - mu, d1 = y1 - mu;
    float rstd = rsqrtf(0.5f * (d0 * d0 + d1 * d1) + 1e-5f);
    float o0 = d0 * rstd * g[0] + b[0];
    float o1 = d1 * rstd * g[1] + b[1];
    if (resid) { o0 += resid[(size_t)i * 2]; o1 += resid[(size_t)i * 2 + 1]; }
    out[(size_t)i * 2] = o0;
    out[(size_t)i * 2 + 1] = o1;
}

// Build EP input [E, 544] bf16 = [x_bf[src] | x_bf[dst] | bf16(e) | 0-pad]
__global__ void build_ep_k(const unsigned short* __restrict__ x_bf,
                           const float* __restrict__ e,
                           const long long* __restrict__ ei,
                           unsigned short* __restrict__ C, long long total) {
    long long tid = (long long)blockIdx.x * blockDim.x + threadIdx.x;
    if (tid >= total) return;
    long long i = tid / EPPAD;
    int c = (int)(tid % EPPAD);
    unsigned short val;
    if (c < 256) {
        long long s = ei[i];
        val = x_bf[(size_t)s * 256 + c];
    } else if (c < 512) {
        long long d = ei[NEDGE + i];
        val = x_bf[(size_t)d * 256 + (c - 256)];
    } else if (c < 514) {
        val = f32_to_bf16(e[i * 2 + (c - 512)]);
    } else {
        val = 0;
    }
    C[tid] = val;
}

// Build NP input [N, 288] bf16 = [x_bf | bf16(agg) | 0-pad]
__global__ void build_np_k(const unsigned short* __restrict__ x_bf,
                           const float* __restrict__ agg,
                           unsigned short* __restrict__ C, long long total) {
    long long tid = (long long)blockIdx.x * blockDim.x + threadIdx.x;
    if (tid >= total) return;
    long long i = tid / NPPAD;
    int c = (int)(tid % NPPAD);
    unsigned short val;
    if (c < 256)      val = x_bf[i * 256 + c];
    else if (c < 258) val = f32_to_bf16(agg[i * 2 + (c - 256)]);
    else              val = 0;
    C[tid] = val;
}

__global__ void zero_f32_k(float* __restrict__ p, long long n) {
    long long tid = (long long)blockIdx.x * blockDim.x + threadIdx.x;
    if (tid < n) p[tid] = 0.0f;
}

__global__ void scatter_add_k(const float* __restrict__ e,
                              const long long* __restrict__ ei,
                              float* __restrict__ agg) {
    int i = blockIdx.x * blockDim.x + threadIdx.x;
    if (i >= NEDGE) return;
    long long d = ei[NEDGE + i];
    atomicAdd(&agg[d * 2 + 0], e[(size_t)i * 2 + 0]);
    atomicAdd(&agg[d * 2 + 1], e[(size_t)i * 2 + 1]);
}

// Tail: d_out += [ (float)latent_edge_index (2*ELAT) | latent_edge_attr (ELAT) ]
__global__ void tail_k(const long long* __restrict__ lei,
                       const float* __restrict__ lea,
                       float* __restrict__ dst) {
    int i = blockIdx.x * blockDim.x + threadIdx.x;
    if (i < 2 * ELAT)        dst[i] = (float)lei[i];
    else if (i < 3 * ELAT)   dst[i] = lea[i - 2 * ELAT];
}

// ---------------------------------------------------------------------------
// Host side
// ---------------------------------------------------------------------------
extern "C" void kernel_launch(void* const* d_in, const int* in_sizes, int n_in,
                              void* d_out, int out_size, void* d_ws, size_t ws_size,
                              hipStream_t stream) {
    (void)in_sizes; (void)n_in; (void)out_size; (void)ws_size;

    // input layout (pytree-flattened setup_inputs order)
    const float* features = (const float*)d_in[0];
    const float* edge_attr = (const float*)d_in[1];
    const float* lat_edge_attr = (const float*)d_in[2];
    const float* nW0 = (const float*)d_in[3],  *nb0 = (const float*)d_in[4];
    const float* nW1 = (const float*)d_in[5],  *nb1 = (const float*)d_in[6];
    const float* nW2 = (const float*)d_in[7],  *nb2 = (const float*)d_in[8];
    const float* nG  = (const float*)d_in[9],  *nB  = (const float*)d_in[10];
    const float* eW0 = (const float*)d_in[11], *eb0 = (const float*)d_in[12];
    const float* eW1 = (const float*)d_in[13], *eb1 = (const float*)d_in[14];
    const float* eW2 = (const float*)d_in[15], *eb2 = (const float*)d_in[16];
    const float* eG  = (const float*)d_in[17], *eB  = (const float*)d_in[18];
    const float* pW0 = (const float*)d_in[19], *pb0 = (const float*)d_in[20];
    const float* pW1 = (const float*)d_in[21], *pb1 = (const float*)d_in[22];
    const float* pW2 = (const float*)d_in[23], *pb2 = (const float*)d_in[24];
    const float* pG  = (const float*)d_in[25], *pB  = (const float*)d_in[26];
    const float* qW0 = (const float*)d_in[27], *qb0 = (const float*)d_in[28];
    const float* qW1 = (const float*)d_in[29], *qb1 = (const float*)d_in[30];
    const float* qW2 = (const float*)d_in[31], *qb2 = (const float*)d_in[32];
    const float* qG  = (const float*)d_in[33], *qB  = (const float*)d_in[34];
    const long long* edge_index = (const long long*)d_in[35];
    const long long* lat_edge_index = (const long long*)d_in[36];

    float* x_out = (float*)d_out;                       // [N,256]
    float* tail_out = x_out + (size_t)NNODE * 256;      // idx + attr

    // workspace carve-up
    size_t off = 0;
    auto alloc = [&](size_t bytes) -> void* {
        off = (off + 255) & ~(size_t)255;
        void* p = (char*)d_ws + off;
        off += bytes;
        return p;
    };
    auto us = [&](size_t elems) { return (unsigned short*)alloc(elems * 2); };
    auto fl = [&](size_t elems) { return (float*)alloc(elems * 4); };

    unsigned short* nW0p = us((size_t)INPAD * HID);
    unsigned short* nW1p = us((size_t)HID * HID);
    unsigned short* nW2p = us((size_t)HID * HID);
    unsigned short* eW0p = us((size_t)EAPAD * HID);
    unsigned short* eW1p = us((size_t)HID * HID);
    unsigned short* eW2p = us((size_t)HID * OUT2PAD);
    unsigned short* pW0p = us((size_t)EPPAD * HID);
    unsigned short* pW1p = us((size_t)HID * HID);
    unsigned short* pW2p = us((size_t)HID * OUT2PAD);
    unsigned short* qW0p = us((size_t)NPPAD * HID);
    unsigned short* qW1p = us((size_t)HID * HID);
    unsigned short* qW2p = us((size_t)HID * HID);

    unsigned short* Xbf  = us((size_t)NNODE * INPAD);
    unsigned short* Ebf  = us((size_t)NEDGE * EAPAD);
    unsigned short* H1   = us((size_t)NNODE * HID);
    unsigned short* H2   = us((size_t)NNODE * HID);
    unsigned short* xbf  = us((size_t)NNODE * HID);
    float*          Yf   = fl((size_t)NNODE * HID);
    unsigned short* Cbuf = us((size_t)NEDGE * EPPAD);   // shared: EP then NP concat
    float* e_pre  = fl((size_t)NEDGE * 2);
    float* e0     = fl((size_t)NEDGE * 2);
    float* ep_pre = fl((size_t)NEDGE * 2);
    float* e1     = fl((size_t)NEDGE * 2);
    float* agg    = fl((size_t)NNODE * 2);

    auto packW = [&](const float* W, int K, int Nout, unsigned short* Wp,
                     int Kpad, int Npad) {
        long long total = (long long)(Npad / 16) * (Kpad / 32) * 512;
        pack_weight_k<<<(unsigned)((total + 255) / 256), 256, 0, stream>>>(
            W, K, Nout, Wp, Kpad / 32, total);
    };
    auto packA = [&](const float* src, unsigned short* dst, long long rows,
                     int K, int Kpad) {
        long long total = rows * Kpad;
        pack_rows_k<<<(unsigned)((total + 255) / 256), 256, 0, stream>>>(
            src, dst, K, Kpad, total);
    };
    auto gemm = [&](const unsigned short* A, int lda, int M,
                    const unsigned short* Wp, int Kpad, const float* bias,
                    int Nout, int Npad, unsigned short* obf, int ldobf,
                    float* of, int ldof, int relu) {
        // each block covers 8 waves * MT tiles * 16 rows = 512 rows
        dim3 grid((unsigned)((M + (128 * MT) - 1) / (128 * MT)),
                  (unsigned)(Npad / 16));
        gemm_bf16_k<<<grid, 256, 0, stream>>>(A, lda, M, Wp, Kpad / 32, bias,
                                              Nout, obf, ldobf, of, ldof, relu);
    };

    // ---- weight + activation packing -------------------------------------
    packW(nW0, INDIM, HID, nW0p, INPAD, HID);
    packW(nW1, HID, HID, nW1p, HID, HID);
    packW(nW2, HID, HID, nW2p, HID, HID);
    packW(eW0, 2, HID, eW0p, EAPAD, HID);
    packW(eW1, HID, HID, eW1p, HID, HID);
    packW(eW2, HID, 2, eW2p, HID, OUT2PAD);
    packW(pW0, 514, HID, pW0p, EPPAD, HID);
    packW(pW1, HID, HID, pW1p, HID, HID);
    packW(pW2, HID, 2, pW2p, HID, OUT2PAD);
    packW(qW0, 258, HID, qW0p, NPPAD, HID);
    packW(qW1, HID, HID, qW1p, HID, HID);
    packW(qW2, HID, HID, qW2p, HID, HID);
    packA(features, Xbf, NNODE, INDIM, INPAD);
    packA(edge_attr, Ebf, NEDGE, 2, EAPAD);

    // ---- node encoder: x = LN(MLP(features)) ------------------------------
    gemm(Xbf, INPAD, NNODE, nW0p, INPAD, nb0, HID, HID, H1, HID, nullptr, 0, 1);
    gemm(H1, HID, NNODE, nW1p, HID, nb1, HID, HID, H2, HID, nullptr, 0, 1);
    gemm(H2, HID, NNODE, nW2p, HID, nb2, HID, HID, nullptr, 0, Yf, HID, 0);
    ln256_k<<<(unsigned)((NNODE + 7) / 8), 256, 0, stream>>>(
        Yf, nG, nB, nullptr, x_out, xbf, NNODE);

    // ---- edge encoder: e0 = LN(MLP(edge_attr)) ----------------------------
    gemm(Ebf, EAPAD, NEDGE, eW0p, EAPAD, eb0, HID, HID, H1, HID, nullptr, 0, 1);
    gemm(H1, HID, NEDGE, eW1p, HID, eb1, HID, HID, H2, HID, nullptr, 0, 1);
    gemm(H2, HID, NEDGE, eW2p, HID, eb2, 2, OUT2PAD, nullptr, 0, e_pre, 2, 0);
    ln2_k<<<(NEDGE + 255) / 256, 256, 0, stream>>>(e_pre, eG, eB, nullptr, e0, NEDGE);

    // ---- edge processor: e1 = LN(MLP([x_src,x_dst,e0])) + e0 --------------
    {
        long long total = (long long)NEDGE * EPPAD;
        build_ep_k<<<(unsigned)((total + 255) / 256), 256, 0, stream>>>(
            xbf, e0, edge_index, Cbuf, total);
    }
    gemm(Cbuf, EPPAD, NEDGE, pW0p, EPPAD, pb0, HID, HID, H1, HID, nullptr, 0, 1);
    gemm(H1, HID, NEDGE, pW1p, HID, pb1, HID, HID, H2, HID, nullptr, 0, 1);
    gemm(H2, HID, NEDGE, pW2p, HID, pb2, 2, OUT2PAD, nullptr, 0, ep_pre, 2, 0);
    ln2_k<<<(NEDGE + 255) / 256, 256, 0, stream>>>(ep_pre, pG, pB, e0, e1, NEDGE);

    // ---- aggregate: agg = segment_sum(e1, dst) ----------------------------
    zero_f32_k<<<(unsigned)(((long long)NNODE * 2 + 255) / 256), 256, 0, stream>>>(
        agg, (long long)NNODE * 2);
    scatter_add_k<<<(NEDGE + 255) / 256, 256, 0, stream>>>(e1, edge_index, agg);

    // ---- node processor: x = LN(MLP([x,agg])) + x -------------------------
    {
        long long total = (long long)NNODE * NPPAD;
        build_np_k<<<(unsigned)((total + 255) / 256), 256, 0, stream>>>(
            xbf, agg, Cbuf, total);
    }
    gemm(Cbuf, NPPAD, NNODE, qW0p, NPPAD, qb0, HID, HID, H1, HID, nullptr, 0, 1);
    gemm(H1, HID, NNODE, qW1p, HID, qb1, HID, HID, H2, HID, nullptr, 0, 1);
    gemm(H2, HID, NNODE, qW2p, HID, qb2, HID, HID, nullptr, 0, Yf, HID, 0);
    ln256_k<<<(unsigned)((NNODE + 7) / 8), 256, 0, stream>>>(
        Yf, qG, qB, x_out, x_out, nullptr, NNODE);

    // ---- tail outputs ------------------------------------------------------
    tail_k<<<(3 * ELAT + 255) / 256, 256, 0, stream>>>(
        lat_edge_index, lat_edge_attr, tail_out);
}
